// WCT_69303592289008
// MI455X (gfx1250) — matrix operand order repaired and checked by reading, not compiled, not run on hardware
//
#include <hip/hip_runtime.h>

// ---------------- CDNA5 WMMA types ----------------
typedef __attribute__((ext_vector_type(16))) __bf16          bf16x16;
typedef __attribute__((ext_vector_type(8)))  float           f32x8;
typedef __attribute__((ext_vector_type(8)))  unsigned short  u16x8;
typedef __attribute__((ext_vector_type(16))) unsigned short  u16x16;

// Native gfx1250 bf16 convert (lowers to v_cvt*_bf16_f32; RNE) instead of bit-twiddling
__device__ __forceinline__ unsigned short f32_to_bf16u(float f) {
    __bf16 h = (__bf16)f;
    return __builtin_bit_cast(unsigned short, h);
}

// A fragment: 16x32 bf16. lanes 0-15: row M=lane, K {0..7,16..23}; lanes 16-31: row M=lane-16, K {8..15,24..31}
__device__ __forceinline__ bf16x16 frag_a(const unsigned short* rowPtr, int lane) {
    int kb = (lane >> 4) * 8;
    u16x8 lo = *(const u16x8*)(rowPtr + kb);
    u16x8 hi = *(const u16x8*)(rowPtr + kb + 16);
    u16x16 v = __builtin_shufflevector(lo, hi, 0,1,2,3,4,5,6,7,8,9,10,11,12,13,14,15);
    return __builtin_bit_cast(bf16x16, v);
}

// B fragment: 32x16 bf16. lane L: col N=L%16; lanes 0-15 hold K {0..7,16..23}, lanes 16-31 hold K {8..15,24..31}
__device__ __forceinline__ bf16x16 frag_b(const unsigned short* colPtr, int lane) {
    int kb = (lane >> 4) * 8;
    u16x8 lo = *(const u16x8*)(colPtr + kb);
    u16x8 hi = *(const u16x8*)(colPtr + kb + 16);
    u16x16 v = __builtin_shufflevector(lo, hi, 0,1,2,3,4,5,6,7,8,9,10,11,12,13,14,15);
    return __builtin_bit_cast(bf16x16, v);
}

// Stage a 64x32 f32 tile of row-major X (rows row0.., cols k0..k0+31) into LDS as bf16 S[64][32]
__device__ __forceinline__ void stage_rows(const float* X, long ld, long row0, long k0,
                                           unsigned short (*S)[32], int tid) {
    int r  = tid >> 2;           // 0..63
    int kc = (tid & 3) * 8;      // 0,8,16,24
    const float* p = X + (row0 + r) * ld + k0 + kc;
    float4 v0 = *(const float4*)p;
    float4 v1 = *(const float4*)(p + 4);
    unsigned short* d = &S[r][kc];
    d[0]=f32_to_bf16u(v0.x); d[1]=f32_to_bf16u(v0.y); d[2]=f32_to_bf16u(v0.z); d[3]=f32_to_bf16u(v0.w);
    d[4]=f32_to_bf16u(v1.x); d[5]=f32_to_bf16u(v1.y); d[6]=f32_to_bf16u(v1.z); d[7]=f32_to_bf16u(v1.w);
}

// Stage a 32x64 f32 tile of row-major B (rows k0..k0+31, cols n0..n0+63) transposed into LDS S[n][k]
__device__ __forceinline__ void stage_cols(const float* B, long ld, long k0, long n0,
                                           unsigned short (*S)[32], int tid) {
    int kk = tid >> 3;           // 0..31
    int nc = (tid & 7) * 8;      // 0..56
    const float* p = B + (k0 + kk) * ld + n0 + nc;
    float4 v0 = *(const float4*)p;
    float4 v1 = *(const float4*)(p + 4);
    S[nc+0][kk]=f32_to_bf16u(v0.x); S[nc+1][kk]=f32_to_bf16u(v0.y);
    S[nc+2][kk]=f32_to_bf16u(v0.z); S[nc+3][kk]=f32_to_bf16u(v0.w);
    S[nc+4][kk]=f32_to_bf16u(v1.x); S[nc+5][kk]=f32_to_bf16u(v1.y);
    S[nc+6][kk]=f32_to_bf16u(v1.z); S[nc+7][kk]=f32_to_bf16u(v1.w);
}

// ---------------- Split-K Gram: Gp[z] += X(i-block) @ X(j-block)^T over K slice z ----------------
// grid (4,4,KSPLIT): 512 blocks -> enough parallelism to saturate HBM at 23.3 TB/s
__global__ __launch_bounds__(256) void wct_gram_part(const float* __restrict__ X, long N, int ksplit,
                                                     float* __restrict__ Gp) {
    __shared__ unsigned short As[64][32];
    __shared__ unsigned short Bs[64][32];
    int tid = threadIdx.x, lane = tid & 31, wv = tid >> 5;
    int wm = wv >> 1, wn = wv & 1;                       // 4x2 waves, each 16x32 of the 64x64 tile
    long i0 = (long)blockIdx.y * 64, j0 = (long)blockIdx.x * 64;
    long kChunk = N / ksplit;
    long kBeg = (long)blockIdx.z * kChunk;
    long kEnd = kBeg + kChunk;
    f32x8 acc0 = {0,0,0,0,0,0,0,0}, acc1 = {0,0,0,0,0,0,0,0};
    for (long k0 = kBeg; k0 < kEnd; k0 += 32) {
        __syncthreads();
        stage_rows(X, N, i0, k0, As, tid);
        stage_rows(X, N, j0, k0, Bs, tid);               // B rows come from X rows too (Gram)
        if (k0 + 32 < kEnd) {
            __builtin_prefetch(X + (i0 + (tid >> 2)) * N + k0 + 32 + (tid & 3) * 8, 0, 1);
            __builtin_prefetch(X + (j0 + (tid >> 2)) * N + k0 + 32 + (tid & 3) * 8, 0, 1);
        }
        __syncthreads();
        bf16x16 a  = frag_a(&As[wm * 16 + (lane & 15)][0], lane);
        bf16x16 b0 = frag_b(&Bs[wn * 32 + (lane & 15)][0], lane);
        bf16x16 b1 = frag_b(&Bs[wn * 32 + 16 + (lane & 15)][0], lane);
        acc0 = __builtin_amdgcn_wmma_f32_16x16x32_bf16(false, a, false, b0, (short)0, acc0, false, false);
        acc1 = __builtin_amdgcn_wmma_f32_16x16x32_bf16(false, a, false, b1, (short)0, acc1, false, false);
    }
    float* Gz = Gp + (long)blockIdx.z * 256 * 256;
    int rowBase = (int)i0 + wm * 16 + ((lane >= 16) ? 8 : 0);
    int col0    = (int)j0 + wn * 32 + (lane & 15);
    for (int r = 0; r < 8; ++r) {
        Gz[(long)(rowBase + r) * 256 + col0]      = acc0[r];
        Gz[(long)(rowBase + r) * 256 + col0 + 16] = acc1[r];
    }
}

// Deterministic split-K reduction: G = sum_z Gp[z]
__global__ __launch_bounds__(256) void wct_gram_reduce(const float* __restrict__ Gp, int ksplit,
                                                       float* __restrict__ G) {
    long idx = (long)blockIdx.x * 256 + threadIdx.x;
    float s = 0.0f;
    for (int z = 0; z < ksplit; ++z) s += Gp[(long)z * 256 * 256 + idx];
    G[idx] = s;
}

// ---------------- Small GEMM: D = eff * A@B + diagAdd*I  (256x256x256) ----------------
// eff = alpha * (numPtr ? sqrt(*numPtr) : 1) * (denPtr ? 1/sqrt(*denPtr) : 1)
__global__ __launch_bounds__(256) void wct_gemm256(const float* __restrict__ A, const float* __restrict__ B,
                                                   float* __restrict__ D, float alpha, float diagAdd,
                                                   const float* __restrict__ numPtr,
                                                   const float* __restrict__ denPtr) {
    __shared__ unsigned short As[64][32];
    __shared__ unsigned short Bs[64][32];
    int tid = threadIdx.x, lane = tid & 31, wv = tid >> 5;
    int wm = wv >> 1, wn = wv & 1;
    long i0 = (long)blockIdx.y * 64, j0 = (long)blockIdx.x * 64;
    f32x8 acc0 = {0,0,0,0,0,0,0,0}, acc1 = {0,0,0,0,0,0,0,0};
    for (long k0 = 0; k0 < 256; k0 += 32) {
        __syncthreads();
        stage_rows(A, 256, i0, k0, As, tid);
        stage_cols(B, 256, k0, j0, Bs, tid);
        __syncthreads();
        bf16x16 a  = frag_a(&As[wm * 16 + (lane & 15)][0], lane);
        bf16x16 b0 = frag_b(&Bs[wn * 32 + (lane & 15)][0], lane);
        bf16x16 b1 = frag_b(&Bs[wn * 32 + 16 + (lane & 15)][0], lane);
        acc0 = __builtin_amdgcn_wmma_f32_16x16x32_bf16(false, a, false, b0, (short)0, acc0, false, false);
        acc1 = __builtin_amdgcn_wmma_f32_16x16x32_bf16(false, a, false, b1, (short)0, acc1, false, false);
    }
    float eff = alpha;
    if (numPtr) eff *= sqrtf(numPtr[0]);
    if (denPtr) eff *= rsqrtf(denPtr[0]);
    int rowBase = (int)i0 + wm * 16 + ((lane >= 16) ? 8 : 0);
    int col0    = (int)j0 + wn * 32 + (lane & 15);
    for (int r = 0; r < 8; ++r) {
        int row = rowBase + r;
        float v0 = eff * acc0[r] + ((row == col0)      ? diagAdd : 0.0f);
        float v1 = eff * acc1[r] + ((row == col0 + 16) ? diagAdd : 0.0f);
        D[(long)row * 256 + col0]      = v0;
        D[(long)row * 256 + col0 + 16] = v1;
    }
}

// ---------------- Apply: out = a*(M@cF + bias) + (1-a)*cF   (256 x N, K=256) ----------------
__global__ __launch_bounds__(256) void wct_apply(const float* __restrict__ Mx, const float* __restrict__ cF,
                                                 const float* __restrict__ bias,
                                                 const float* __restrict__ alphaPtr,
                                                 float* __restrict__ out, long N) {
    __shared__ unsigned short As[64][32];
    __shared__ unsigned short Bs[64][32];
    int tid = threadIdx.x, lane = tid & 31, wv = tid >> 5;
    int wm = wv >> 1, wn = wv & 1;
    long i0 = (long)blockIdx.y * 64;                     // 4 row-blocks
    long j0 = (long)blockIdx.x * 64;                     // N/64 col-blocks
    f32x8 acc0 = {0,0,0,0,0,0,0,0}, acc1 = {0,0,0,0,0,0,0,0};
    for (long k0 = 0; k0 < 256; k0 += 32) {
        __syncthreads();
        stage_rows(Mx, 256, i0, k0, As, tid);
        stage_cols(cF, N, k0, j0, Bs, tid);
        __syncthreads();
        bf16x16 a  = frag_a(&As[wm * 16 + (lane & 15)][0], lane);
        bf16x16 b0 = frag_b(&Bs[wn * 32 + (lane & 15)][0], lane);
        bf16x16 b1 = frag_b(&Bs[wn * 32 + 16 + (lane & 15)][0], lane);
        acc0 = __builtin_amdgcn_wmma_f32_16x16x32_bf16(false, a, false, b0, (short)0, acc0, false, false);
        acc1 = __builtin_amdgcn_wmma_f32_16x16x32_bf16(false, a, false, b1, (short)0, acc1, false, false);
    }
    float al  = alphaPtr[0];
    float bl  = 1.0f - al;
    int rowBase = (int)i0 + wm * 16 + ((lane >= 16) ? 8 : 0);
    long col0   = j0 + wn * 32 + (lane & 15);
    for (int r = 0; r < 8; ++r) {
        int  row  = rowBase + r;
        long idx0 = (long)row * N + col0;
        long idx1 = idx0 + 16;
        float t0 = acc0[r] + bias[row];
        float t1 = acc1[r] + bias[row];
        out[idx0] = al * t0 + bl * cF[idx0];
        out[idx1] = al * t1 + bl * cF[idx1];
    }
}

// ---------------- Row sums (for means) ----------------
__global__ __launch_bounds__(256) void wct_rowsum(const float* __restrict__ X, long N,
                                                  float* __restrict__ S) {
    __shared__ float red[256];
    long row = blockIdx.x;
    const float* p = X + row * N;
    float s = 0.0f;
    for (long k = (long)threadIdx.x * 4; k < N; k += 1024) {
        float4 v = *(const float4*)(p + k);
        s += v.x + v.y + v.z + v.w;
    }
    red[threadIdx.x] = s;
    __syncthreads();
    for (int off = 128; off > 0; off >>= 1) {
        if ((int)threadIdx.x < off) red[threadIdx.x] += red[threadIdx.x + off];
        __syncthreads();
    }
    if (threadIdx.x == 0) S[row] = red[0];
}

// ---------------- Cov fixup: Cov = (G - S S^T / N)/(N-1) + diag*I ----------------
__global__ __launch_bounds__(256) void wct_finalize(const float* __restrict__ G, const float* __restrict__ S,
                                                    float* __restrict__ Cov, float diag,
                                                    float invN, float invNm1) {
    int i = blockIdx.x, j = threadIdx.x;
    float v = (G[(long)i * 256 + j] - S[i] * S[j] * invN) * invNm1;
    if (i == j) v += diag;
    Cov[(long)i * 256 + j] = v;
}

// ---------------- Trace of 256x256 ----------------
__global__ __launch_bounds__(256) void wct_trace(const float* __restrict__ A, float* __restrict__ outv) {
    __shared__ float red[256];
    red[threadIdx.x] = A[(long)threadIdx.x * 257];
    __syncthreads();
    for (int off = 128; off > 0; off >>= 1) {
        if ((int)threadIdx.x < off) red[threadIdx.x] += red[threadIdx.x + off];
        __syncthreads();
    }
    if (threadIdx.x == 0) outv[0] = red[0];
}

// ---------------- Newton-Schulz init: Y = Cov/tr, Z = I ----------------
__global__ __launch_bounds__(256) void wct_nsinit(const float* __restrict__ Cov, const float* __restrict__ tr,
                                                  float* __restrict__ Y, float* __restrict__ Z) {
    int idx = blockIdx.x * 256 + threadIdx.x;
    float invt = 1.0f / tr[0];
    Y[idx] = Cov[idx] * invt;
    Z[idx] = ((idx >> 8) == (idx & 255)) ? 1.0f : 0.0f;
}

// ---------------- bias = s_mean - M @ c_mean ----------------
__global__ __launch_bounds__(256) void wct_bias(const float* __restrict__ Mx, const float* __restrict__ Sc,
                                                const float* __restrict__ Ss, float* __restrict__ bias,
                                                float invNc, float invNs) {
    __shared__ float cm[256];
    int i = threadIdx.x;
    cm[i] = Sc[i] * invNc;
    __syncthreads();
    float dot = 0.0f;
    const float* row = Mx + (long)i * 256;
    for (int j = 0; j < 256; ++j) dot += row[j] * cm[j];
    bias[i] = Ss[i] * invNs - dot;
}

// ---------------- Host orchestration ----------------
extern "C" void kernel_launch(void* const* d_in, const int* in_sizes, int n_in,
                              void* d_out, int out_size, void* d_ws, size_t ws_size,
                              hipStream_t stream) {
    (void)n_in; (void)out_size; (void)ws_size;
    const float* cF     = (const float*)d_in[0];
    const float* sF     = (const float*)d_in[1];
    const float* alphaP = (const float*)d_in[2];
    float* out = (float*)d_out;
    float* ws  = (float*)d_ws;

    const long N = (long)in_sizes[0] / 256;   // 65536 flattened spatial
    const long MSZ = 256 * 256;
    const int  KSPLIT = 32;                   // 512 blocks per Gram pass

    float* Gc   = ws +  0 * MSZ;
    float* Gs   = ws +  1 * MSZ;
    float* Covc = ws +  2 * MSZ;
    float* Covs = ws +  3 * MSZ;
    float* Yc0  = ws +  4 * MSZ;  float* Yc1 = ws +  5 * MSZ;
    float* Zc0  = ws +  6 * MSZ;  float* Zc1 = ws +  7 * MSZ;
    float* Tc   = ws +  8 * MSZ;
    float* Ys0  = ws +  9 * MSZ;  float* Ys1 = ws + 10 * MSZ;
    float* Zs0  = ws + 11 * MSZ;  float* Zs1 = ws + 12 * MSZ;
    float* Ts   = ws + 13 * MSZ;
    float* Mw   = ws + 14 * MSZ;
    float* Sc   = ws + 15 * MSZ;
    float* Ss   = Sc + 256;
    float* trc  = Ss + 256;
    float* trs  = trc + 1;
    float* bias = trs + 1;
    float* Gp   = ws + 16 * MSZ;              // KSPLIT * 256*256 partials (8 MB)

    dim3 blk(256);
    dim3 g44(4, 4);
    dim3 gGram(4, 4, KSPLIT);
    const float invN   = 1.0f / (float)N;
    const float invNm1 = 1.0f / (float)(N - 1);

    // Streaming Gram (split-K for HBM saturation) + row sums
    wct_gram_part<<<gGram, blk, 0, stream>>>(cF, N, KSPLIT, Gp);
    wct_gram_reduce<<<256, blk, 0, stream>>>(Gp, KSPLIT, Gc);
    wct_gram_part<<<gGram, blk, 0, stream>>>(sF, N, KSPLIT, Gp);
    wct_gram_reduce<<<256, blk, 0, stream>>>(Gp, KSPLIT, Gs);
    wct_rowsum<<<256, blk, 0, stream>>>(cF, N, Sc);
    wct_rowsum<<<256, blk, 0, stream>>>(sF, N, Ss);

    // Covariances (mean correction folded as rank-1 fixup), +I for content
    wct_finalize<<<256, blk, 0, stream>>>(Gc, Sc, Covc, 1.0f, invN, invNm1);
    wct_finalize<<<256, blk, 0, stream>>>(Gs, Ss, Covs, 0.0f, invN, invNm1);
    wct_trace<<<1, blk, 0, stream>>>(Covc, trc);
    wct_trace<<<1, blk, 0, stream>>>(Covs, trs);

    // Newton-Schulz: Z -> Cov^{-1/2}*sqrt(tr), Y -> Cov^{1/2}/sqrt(tr)
    wct_nsinit<<<256, blk, 0, stream>>>(Covc, trc, Yc0, Zc0);
    wct_nsinit<<<256, blk, 0, stream>>>(Covs, trs, Ys0, Zs0);

    const int NITER = 16;
    float *Y = Yc0, *Yn = Yc1, *Z = Zc0, *Zn = Zc1;
    for (int it = 0; it < NITER; ++it) {
        wct_gemm256<<<g44, blk, 0, stream>>>(Z, Y, Tc, -0.5f, 1.5f, nullptr, nullptr); // T = (3I - ZY)/2
        wct_gemm256<<<g44, blk, 0, stream>>>(Y, Tc, Yn, 1.0f, 0.0f, nullptr, nullptr); // Y' = Y T
        wct_gemm256<<<g44, blk, 0, stream>>>(Tc, Z, Zn, 1.0f, 0.0f, nullptr, nullptr); // Z' = T Z
        float* t = Y; Y = Yn; Yn = t;
        t = Z; Z = Zn; Zn = t;
    }
    float* ZcFinal = Z;   // content whitening direction

    Y = Ys0; Yn = Ys1; Z = Zs0; Zn = Zs1;
    for (int it = 0; it < NITER; ++it) {
        wct_gemm256<<<g44, blk, 0, stream>>>(Z, Y, Ts, -0.5f, 1.5f, nullptr, nullptr);
        wct_gemm256<<<g44, blk, 0, stream>>>(Y, Ts, Yn, 1.0f, 0.0f, nullptr, nullptr);
        wct_gemm256<<<g44, blk, 0, stream>>>(Ts, Z, Zn, 1.0f, 0.0f, nullptr, nullptr);
        float* t = Y; Y = Yn; Yn = t;
        t = Z; Z = Zn; Zn = t;
    }
    float* YsFinal = Y;   // style coloring direction

    // M = sqrt(trs/trc) * Ys @ Zc  (combined color∘whiten operator)
    wct_gemm256<<<g44, blk, 0, stream>>>(YsFinal, ZcFinal, Mw, 1.0f, 0.0f, trs, trc);

    // bias = s_mean - M @ c_mean
    wct_bias<<<1, blk, 0, stream>>>(Mw, Sc, Ss, bias, invN, invN);

    // out = alpha*(M@cF + bias) + (1-alpha)*cF
    dim3 gApply((unsigned)(N / 64), 4);
    wct_apply<<<gApply, blk, 0, stream>>>(Mw, cF, bias, alphaP, out, N);
}
